// FlowNet3D_GlobalAttention_7267084665325
// MI455X (gfx1250) — compile-verified, matrix operand order checked by reading, not run on hardware
//
// FlowNet3D + point-transformer forward pass for MI455X (gfx1250, wave32, WMMA).
//
// Design notes (compile-only; reasoning from MI455X specs):
//  * Every matmul in the network (SA/upconv MLPs, pt_block pairwise GEMMs,
//    flow attention) is routed through v_wmma_f32_16x16x32_f16: f16 A/B with
//    f32 accumulation. K dims are <= 387, weights (<256KB) stay resident in
//    the 192MB L2, so activation streaming dominates -> f16 operands halve
//    A/B traffic and use the matrix pipe instead of VALU FMAs.
//  * GEMM: 128-thread block (4 waves) computes a 64x64 C tile. A and B are
//    cooperatively staged into LDS as f16 (B transposed to [n][k] so that
//    WMMA fragment K-pairs are contiguous, 4B aligned -> ds_load_b32).
//    Bounds checks live only in the staging step, with an unguarded
//    fast path for interior tiles; the WMMA phase is branch-free.
//    Each wave owns a 16-col strip and 4 M sub-tiles (B-fragment reuse x4,
//    LDS A tile shared by 4 waves): 8 WMMAs per 64-K stage.
//  * Irregular stages (FPS, ball query, KNN, gathers, per-channel softmax,
//    3-NN interpolation) are scalar kernels; wave32 + small blocks give high
//    WGP occupancy to hide latency.
//  * Scratch comes from a bump arena in d_ws. Peak transient use is pt2
//    (three B*256*256*64 f32 tensors ~ 400MB) + ~45MB persistents.
//
// Assumed d_in order: setup_inputs() insertion order: xyzs, points, then
// params in make_params insertion order with lists flattened in order and
// pt dicts in (Wq,Wk,Wv,Wp1,bp1,Wp2,bp2,Wg1,bg1,Wg2,bg2,Wo) order.

#include <hip/hip_runtime.h>
#include <cmath>

#define BB   8
#define NPTS 8192
#define KNN_MAX 32

typedef __attribute__((ext_vector_type(16))) _Float16 v16h;
typedef __attribute__((ext_vector_type(8)))  float    v8f;

static inline int blk(long long total) { return (int)((total + 255) / 256); }

// --------------------------------------------------------------------------
// WMMA GEMM:  C[b,m,n] = act(alpha * sum_k A[b,m,k]*B(k,n) + bias[n]
//                            + res[b,m,n])
// B(k,n) = transB ? Bm[b, n, k] : Bm[b, k, n].
// Block = 128 threads = 4 waves -> 64(M) x 64(N) C tile, K staged 64 at a
// time through LDS in f16.  Wave w owns columns [w*16, w*16+16); loops over
// 4 M sub-tiles reusing its B fragment.
// --------------------------------------------------------------------------
#define GTM 64
#define GTN 64
#define GTK 64
#define LDK (GTK + 8)   // pad to dodge LDS bank conflicts

__global__ __launch_bounds__(128)
void k_gemm_wmma(const float* __restrict__ A, const float* __restrict__ Bm,
                 const float* __restrict__ bias, const float* __restrict__ res,
                 float* __restrict__ C,
                 int M, int Nn, int Kk, int lda, int ldb, int ldc,
                 long long sA, long long sB, long long sC, long long sR,
                 float alpha, int relu, int transB)
{
    __shared__ _Float16 lA[GTM][LDK];
    __shared__ _Float16 lB[GTN][LDK];   // B stored [n][k]

    const int tid  = threadIdx.x;
    const int lane = tid & 31;
    const int wv   = tid >> 5;
    const int m0   = blockIdx.x * GTM;
    const int n0   = blockIdx.y * GTN;
    const int b    = blockIdx.z;
    const float* Ab = A  + (long long)b * sA;
    const float* Bb = Bm + (long long)b * sB;

    const int l15 = lane & 15;
    const int kh  = lane >> 4;          // which K-half this lane holds

    v8f acc0 = {}, acc1 = {}, acc2 = {}, acc3 = {};

    for (int k0 = 0; k0 < Kk; k0 += GTK) {
        const bool full = (m0 + GTM <= M) && (n0 + GTN <= Nn) && (k0 + GTK <= Kk);
        if (full) {
            // -------- unguarded staging (interior tiles) ------------------
            for (int i = tid; i < GTM * GTK; i += 128) {
                const int r = i >> 6, k = i & (GTK - 1);
                lA[r][k] = (_Float16)Ab[(long long)(m0 + r) * lda + (k0 + k)];
            }
            if (transB) {
                for (int i = tid; i < GTN * GTK; i += 128) {
                    const int n = i >> 6, k = i & (GTK - 1);
                    lB[n][k] = (_Float16)Bb[(long long)(n0 + n) * ldb + (k0 + k)];
                }
            } else {
                for (int i = tid; i < GTN * GTK; i += 128) {
                    const int n = i >> 6, k = i & (GTK - 1);
                    lB[n][k] = (_Float16)Bb[(long long)(k0 + k) * ldb + (n0 + n)];
                }
            }
        } else {
            // -------- guarded staging (edge tiles), zero-fill OOB ---------
            for (int i = tid; i < GTM * GTK; i += 128) {
                const int r = i >> 6, k = i & (GTK - 1);
                const int gr = m0 + r, gk = k0 + k;
                float v = 0.f;
                if (gr < M && gk < Kk) v = Ab[(long long)gr * lda + gk];
                lA[r][k] = (_Float16)v;
            }
            for (int i = tid; i < GTN * GTK; i += 128) {
                const int n = i >> 6, k = i & (GTK - 1);
                const int gn = n0 + n, gk = k0 + k;
                float v = 0.f;
                if (gn < Nn && gk < Kk)
                    v = transB ? Bb[(long long)gn * ldb + gk]
                               : Bb[(long long)gk * ldb + gn];
                lB[n][k] = (_Float16)v;
            }
        }
        __syncthreads();

        // -------- branch-free WMMA phase: 2 K-substeps x 4 M sub-tiles ----
        const int colL = wv * 16 + l15;
#pragma unroll
        for (int ks = 0; ks < 2; ++ks) {
            v16h bf;
#pragma unroll
            for (int v = 0; v < 8; ++v) {
                // CDNA5 16-bit 16x32 fragment layout:
                // K base = 32*ks + 8*kh + (v>=4 ? 16 : 0) + 2*(v&3)
                const int kb = ks * 32 + kh * 8 + ((v & 4) ? 16 : 0) + (v & 3) * 2;
                bf[2 * v]     = lB[colL][kb];
                bf[2 * v + 1] = lB[colL][kb + 1];
            }
#pragma unroll
            for (int mt = 0; mt < 4; ++mt) {
                v16h af;
                const int rowL = mt * 16 + l15;
#pragma unroll
                for (int v = 0; v < 8; ++v) {
                    const int kb = ks * 32 + kh * 8 + ((v & 4) ? 16 : 0) + (v & 3) * 2;
                    af[2 * v]     = lA[rowL][kb];
                    af[2 * v + 1] = lA[rowL][kb + 1];
                }
                v8f& acc = (mt == 0) ? acc0 : (mt == 1) ? acc1 : (mt == 2) ? acc2 : acc3;
                acc = __builtin_amdgcn_wmma_f32_16x16x32_f16(
                    false, af, false, bf, (short)0, acc, false, false);
            }
        }
        __syncthreads();
    }

    // ---------------- epilogue -------------------------------------------
    const int col = n0 + wv * 16 + l15;
    if (col < Nn) {
        const float bv = bias ? bias[col] : 0.f;
#pragma unroll
        for (int mt = 0; mt < 4; ++mt) {
            const v8f& acc = (mt == 0) ? acc0 : (mt == 1) ? acc1 : (mt == 2) ? acc2 : acc3;
#pragma unroll
            for (int r = 0; r < 8; ++r) {
                const int rr = m0 + mt * 16 + r + 8 * kh;   // D-matrix layout
                if (rr < M) {
                    float v = acc[r] * alpha + bv;
                    if (res) v += res[(long long)b * sR + (long long)rr * ldc + col];
                    if (relu && v < 0.f) v = 0.f;
                    C[(long long)b * sC + (long long)rr * ldc + col] = v;
                }
            }
        }
    }
}

// --------------------------------------------------------------------------
// Data movement / pointnet primitives
// --------------------------------------------------------------------------

// (B, C, Np) channel-first (with per-batch stride) -> (B, Np, C) channel-last
__global__ void k_cf_to_cl(const float* __restrict__ src, float* __restrict__ dst,
                           int Bn, int C, int Np, long long srcBatchStride)
{
    long long t = (long long)blockIdx.x * blockDim.x + threadIdx.x;
    long long total = (long long)Bn * C * Np;
    if (t >= total) return;
    int n = (int)(t % Np);
    int c = (int)((t / Np) % C);
    int b = (int)(t / ((long long)C * Np));
    dst[((long long)b * Np + n) * C + c] = src[(long long)b * srcBatchStride + (long long)c * Np + n];
}

// Farthest point sampling: one 256-thread block per batch.
__global__ __launch_bounds__(256)
void k_fps(const float* __restrict__ xyz, int* __restrict__ out,
           float* __restrict__ distws, int Np, int S)
{
    const int b = blockIdx.x, tid = threadIdx.x;
    const float* xb = xyz + (long long)b * Np * 3;
    float* dist = distws + (long long)b * Np;
    __shared__ float sv[256];
    __shared__ int   si[256];
    __shared__ int   sfar;
    for (int i = tid; i < Np; i += 256) dist[i] = 1e10f;
    if (tid == 0) sfar = 0;
    __syncthreads();
    for (int s = 0; s < S; ++s) {
        const int far = sfar;
        if (tid == 0) out[b * S + s] = far;
        const float cx = xb[far * 3], cy = xb[far * 3 + 1], cz = xb[far * 3 + 2];
        float bv = -1.f; int bi = 0x7fffffff;
        for (int i = tid; i < Np; i += 256) {
            const float dx = xb[i * 3] - cx, dy = xb[i * 3 + 1] - cy, dz = xb[i * 3 + 2] - cz;
            const float d  = dx * dx + dy * dy + dz * dz;
            const float nd = fminf(dist[i], d);
            dist[i] = nd;
            if (nd > bv) { bv = nd; bi = i; }
        }
        sv[tid] = bv; si[tid] = bi;
        __syncthreads();
        for (int off = 128; off > 0; off >>= 1) {
            if (tid < off) {
                if (sv[tid + off] > sv[tid] ||
                    (sv[tid + off] == sv[tid] && si[tid + off] < si[tid])) {
                    sv[tid] = sv[tid + off]; si[tid] = si[tid + off];
                }
            }
            __syncthreads();
        }
        if (tid == 0) sfar = si[0];
        __syncthreads();
    }
}

// Ball query: first K indices (ascending) with d <= r^2; pad with first hit.
__global__ void k_ball_query(const float* __restrict__ xyz, const float* __restrict__ nxyz,
                             int* __restrict__ out, int Np, int S, int K, float r2, int Bn)
{
    long long t = (long long)blockIdx.x * blockDim.x + threadIdx.x;
    if (t >= (long long)Bn * S) return;
    const int s = (int)(t % S), b = (int)(t / S);
    const float* c = nxyz + ((long long)b * S + s) * 3;
    const float cx = c[0], cy = c[1], cz = c[2];
    const float* xb = xyz + (long long)b * Np * 3;
    long long base = ((long long)b * S + s) * K;
    int cnt = 0, first = -1;
    for (int i = 0; i < Np && cnt < K; ++i) {
        const float dx = xb[i * 3] - cx, dy = xb[i * 3 + 1] - cy, dz = xb[i * 3 + 2] - cz;
        if (dx * dx + dy * dy + dz * dz <= r2) {
            if (first < 0) first = i;
            out[base + cnt++] = i;
        }
    }
    if (first < 0) first = Np - 1;
    while (cnt < K) out[base + cnt++] = first;
}

// KNN (k <= 32), ascending distance, ties keep lower index.
__global__ void k_knn(const float* __restrict__ q, const float* __restrict__ pts,
                      int* __restrict__ idx, int S, int Mp, int K, int Bn)
{
    long long t = (long long)blockIdx.x * blockDim.x + threadIdx.x;
    if (t >= (long long)Bn * S) return;
    const int s = (int)(t % S), b = (int)(t / S);
    const float* c = q + ((long long)b * S + s) * 3;
    const float cx = c[0], cy = c[1], cz = c[2];
    const float* xb = pts + (long long)b * Mp * 3;
    float bd[KNN_MAX]; int bi[KNN_MAX];
    for (int j = 0; j < K; ++j) { bd[j] = 1e30f; bi[j] = 0; }
    for (int m = 0; m < Mp; ++m) {
        const float dx = xb[m * 3] - cx, dy = xb[m * 3 + 1] - cy, dz = xb[m * 3 + 2] - cz;
        const float d = dx * dx + dy * dy + dz * dz;
        if (d < bd[K - 1]) {
            int p = K - 1;
            while (p > 0 && d < bd[p - 1]) { bd[p] = bd[p - 1]; bi[p] = bi[p - 1]; --p; }
            bd[p] = d; bi[p] = m;
        }
    }
    long long base = ((long long)b * S + s) * K;
    for (int j = 0; j < K; ++j) idx[base + j] = bi[j];
}

// Gather grouped points into rows of width rowC at channel offset; optional
// per-(b,s) center subtraction. K==1 gathers single points.
__global__ void k_gather_group(const float* __restrict__ pts, const int* __restrict__ idx,
                               const float* __restrict__ center, float* __restrict__ out,
                               int Mp, int S, int K, int C, int rowC, int chOff, int Bn)
{
    long long t = (long long)blockIdx.x * blockDim.x + threadIdx.x;
    long long total = (long long)Bn * S * K * C;
    if (t >= total) return;
    int c = (int)(t % C);
    long long r = t / C;
    int k = (int)(r % K); r /= K;
    int s = (int)(r % S);
    int b = (int)(r / S);
    int id = idx[((long long)b * S + s) * K + k];
    if (id < 0) id = 0;
    if (id >= Mp) id = Mp - 1;
    float v = pts[((long long)b * Mp + id) * C + c];
    if (center) v -= center[((long long)b * S + s) * C + c];
    out[(((long long)b * S + s) * K + k) * (long long)rowC + chOff + c] = v;
}

// Replicate per-row features across K slots at channel offset (K==1: concat).
__global__ void k_scatter_ch(const float* __restrict__ src, float* __restrict__ out,
                             long long rows, int K, int C, int rowC, int chOff)
{
    long long t = (long long)blockIdx.x * blockDim.x + threadIdx.x;
    long long total = rows * K * C;
    if (t >= total) return;
    int c = (int)(t % C);
    long long r = t / C;
    int k = (int)(r % K);
    long long row = r / K;
    out[(row * K + k) * (long long)rowC + chOff + c] = src[row * C + c];
}

__global__ void k_maxpool(const float* __restrict__ in, float* __restrict__ out,
                          long long rows, int K, int C)
{
    long long t = (long long)blockIdx.x * blockDim.x + threadIdx.x;
    if (t >= rows * C) return;
    int c = (int)(t % C);
    long long row = t / C;
    const float* p = in + (row * K) * (long long)C + c;
    float m = p[0];
    for (int k = 1; k < K; ++k) m = fmaxf(m, p[(long long)k * C]);
    out[row * C + c] = m;
}

// rel[b,n,m,:] = x[b,n,:] - x[b,m,:]
__global__ void k_rel(const float* __restrict__ x, float* __restrict__ rel, int Bn, int S)
{
    long long t = (long long)blockIdx.x * blockDim.x + threadIdx.x;
    long long total = (long long)Bn * S * S * 3;
    if (t >= total) return;
    int c = (int)(t % 3);
    long long r = t / 3;
    int m = (int)(r % S); r /= S;
    int n = (int)(r % S);
    int b = (int)(r / S);
    rel[t] = x[((long long)b * S + n) * 3 + c] - x[((long long)b * S + m) * 3 + c];
}

// g[b,n,m,c] = Q[b,n,c] - K[b,m,c] + pos[b,n,m,c]
__global__ void k_build_g(const float* __restrict__ Q, const float* __restrict__ Kt,
                          const float* __restrict__ pos, float* __restrict__ g,
                          int Bn, int S, int C)
{
    long long t = (long long)blockIdx.x * blockDim.x + threadIdx.x;
    long long total = (long long)Bn * S * S * C;
    if (t >= total) return;
    int c = (int)(t % C);
    long long r = t / C;
    int m = (int)(r % S); r /= S;
    int n = (int)(r % S);
    int b = (int)(r / S);
    g[t] = Q[((long long)b * S + n) * C + c] - Kt[((long long)b * S + m) * C + c] + pos[t];
}

// In-place softmax over the m axis of (B,S,S,C): one thread per (b,n,c).
__global__ void k_softmax_mid(float* __restrict__ g, int Bn, int S, int C)
{
    long long t = (long long)blockIdx.x * blockDim.x + threadIdx.x;
    if (t >= (long long)Bn * S * C) return;
    int c = (int)(t % C);
    long long r = t / C;
    int n = (int)(r % S);
    int b = (int)(r / S);
    float* p = g + (((long long)b * S + n) * S) * (long long)C + c;
    float mx = -1e30f;
    for (int m = 0; m < S; ++m) mx = fmaxf(mx, p[(long long)m * C]);
    float sum = 0.f;
    for (int m = 0; m < S; ++m) { float e = expf(p[(long long)m * C] - mx); p[(long long)m * C] = e; sum += e; }
    float inv = 1.f / sum;
    for (int m = 0; m < S; ++m) p[(long long)m * C] *= inv;
}

// out[b,n,c] = sum_m attn[b,n,m,c] * (V[b,m,c] + pos[b,n,m,c])
__global__ void k_attn_out(const float* __restrict__ attn, const float* __restrict__ pos,
                           const float* __restrict__ V, float* __restrict__ out,
                           int Bn, int S, int C)
{
    long long t = (long long)blockIdx.x * blockDim.x + threadIdx.x;
    if (t >= (long long)Bn * S * C) return;
    int c = (int)(t % C);
    long long r = t / C;
    int n = (int)(r % S);
    int b = (int)(r / S);
    const long long pbase = (((long long)b * S + n) * S) * (long long)C + c;
    float acc = 0.f;
    for (int m = 0; m < S; ++m) {
        const long long o = pbase + (long long)m * C;
        acc += attn[o] * (V[((long long)b * S + m) * C + c] + pos[o]);
    }
    out[((long long)b * S + n) * C + c] = acc;
}

// Row-wise softmax over last axis (one thread per row).
__global__ void k_softmax_last(float* __restrict__ a, long long rows, int L)
{
    long long t = (long long)blockIdx.x * blockDim.x + threadIdx.x;
    if (t >= rows) return;
    float* p = a + t * L;
    float mx = -1e30f;
    for (int i = 0; i < L; ++i) mx = fmaxf(mx, p[i]);
    float sum = 0.f;
    for (int i = 0; i < L; ++i) { float e = expf(p[i] - mx); p[i] = e; sum += e; }
    float inv = 1.f / sum;
    for (int i = 0; i < L; ++i) p[i] *= inv;
}

__global__ void k_add(float* __restrict__ dst, const float* __restrict__ a,
                      const float* __restrict__ b, long long n)
{
    long long t = (long long)blockIdx.x * blockDim.x + threadIdx.x;
    if (t < n) dst[t] = a[t] + b[t];
}

// 3-NN inverse-distance interpolation of f2 onto x1, written into rows of
// width rowC at channel offset chOff.
__global__ void k_fp_interp(const float* __restrict__ x1, const float* __restrict__ x2,
                            const float* __restrict__ f2, float* __restrict__ out,
                            int Bn, int Np, int Mp, int C, int rowC, int chOff)
{
    long long t = (long long)blockIdx.x * blockDim.x + threadIdx.x;
    if (t >= (long long)Bn * Np) return;
    int n = (int)(t % Np), b = (int)(t / Np);
    const float* c = x1 + ((long long)b * Np + n) * 3;
    const float cx = c[0], cy = c[1], cz = c[2];
    const float* xb = x2 + (long long)b * Mp * 3;
    float bd[3] = {1e30f, 1e30f, 1e30f};
    int   bi[3] = {0, 0, 0};
    for (int m = 0; m < Mp; ++m) {
        const float dx = xb[m * 3] - cx, dy = xb[m * 3 + 1] - cy, dz = xb[m * 3 + 2] - cz;
        const float d = dx * dx + dy * dy + dz * dz;
        if (d < bd[2]) {
            int p = 2;
            while (p > 0 && d < bd[p - 1]) { bd[p] = bd[p - 1]; bi[p] = bi[p - 1]; --p; }
            bd[p] = d; bi[p] = m;
        }
    }
    float w0 = 1.f / (bd[0] + 1e-8f), w1 = 1.f / (bd[1] + 1e-8f), w2 = 1.f / (bd[2] + 1e-8f);
    const float inv = 1.f / (w0 + w1 + w2);
    w0 *= inv; w1 *= inv; w2 *= inv;
    const float* p0 = f2 + ((long long)b * Mp + bi[0]) * C;
    const float* p1 = f2 + ((long long)b * Mp + bi[1]) * C;
    const float* p2 = f2 + ((long long)b * Mp + bi[2]) * C;
    float* o = out + ((long long)b * Np + n) * rowC + chOff;
    for (int cc = 0; cc < C; ++cc) o[cc] = w0 * p0[cc] + w1 * p1[cc] + w2 * p2[cc];
}

// log_softmax over width-2 rows.
__global__ void k_logsoftmax2(const float* __restrict__ in, float* __restrict__ out, long long rows)
{
    long long t = (long long)blockIdx.x * blockDim.x + threadIdx.x;
    if (t >= rows) return;
    const float a = in[t * 2], b = in[t * 2 + 1];
    const float m = fmaxf(a, b);
    const float lse = m + logf(expf(a - m) + expf(b - m));
    out[t * 2] = a - lse;
    out[t * 2 + 1] = b - lse;
}

// --------------------------------------------------------------------------
// Host-side orchestration
// --------------------------------------------------------------------------
namespace {

struct Arena {
    char*  base;
    size_t off, cap;
    Arena(void* b, size_t c) : base((char*)b), off(0), cap(c) {}
    template <typename T> T* get(size_t n) {
        size_t bytes = (n * sizeof(T) + 255) & ~(size_t)255;
        T* p = (T*)(base + off);
        off += bytes;
        return p;
    }
    size_t mark() const { return off; }
    void   reset(size_t m) { off = m; }
};

void gemm(hipStream_t st, const float* A, const float* Bm, const float* bias,
          const float* res, float* C, int M, int Nn, int Kk, bool relu,
          float alpha = 1.f, bool transB = false, int batch = 1,
          long long sA = 0, long long sB = 0, long long sC = 0, long long sR = 0)
{
    const int lda = Kk;
    const int ldb = transB ? Kk : Nn;
    const int ldc = Nn;
    dim3 g((unsigned)((M + GTM - 1) / GTM), (unsigned)((Nn + GTN - 1) / GTN), (unsigned)batch);
    k_gemm_wmma<<<g, dim3(128, 1, 1), 0, st>>>(A, Bm, bias, res, C, M, Nn, Kk,
                                               lda, ldb, ldc, sA, sB, sC, sR,
                                               alpha, relu ? 1 : 0, transB ? 1 : 0);
}

// Chain of ReLU GEMMs (dims has nw+1 entries).
void mlp_run(hipStream_t st, Arena& ar, const float* in, int rows,
             const float* const* Ws, const int* dims, int nw, float* out)
{
    size_t m0 = ar.mark();
    int maxd = 0;
    for (int i = 1; i < nw; ++i) maxd = dims[i] > maxd ? dims[i] : maxd;
    float* t0 = (nw > 1) ? ar.get<float>((size_t)rows * maxd) : nullptr;
    float* t1 = (nw > 2) ? ar.get<float>((size_t)rows * maxd) : nullptr;
    const float* cur = in;
    for (int i = 0; i < nw; ++i) {
        float* dst = (i == nw - 1) ? out : ((i & 1) ? t1 : t0);
        gemm(st, cur, Ws[i], nullptr, nullptr, dst, rows, dims[i + 1], dims[i], true);
        cur = dst;
    }
    ar.reset(m0);
}

void set_abstraction(hipStream_t st, Arena& ar, const float* x, const float* f,
                     int Min, int Cin, int S, float radius, int K,
                     const float* const* Ws, const int* dims, int nw,
                     float* ox, float* of)
{
    size_t m0 = ar.mark();
    float* dist = ar.get<float>((size_t)BB * Min);
    int*   fidx = ar.get<int>((size_t)BB * S);
    int*   gidx = ar.get<int>((size_t)BB * S * K);
    const int rowC = 3 + Cin;
    const int Cout = dims[nw];
    float* grp  = ar.get<float>((size_t)BB * S * K * rowC);
    float* feat = ar.get<float>((size_t)BB * S * K * Cout);

    k_fps<<<BB, 256, 0, st>>>(x, fidx, dist, Min, S);
    k_gather_group<<<blk((long long)BB * S * 3), 256, 0, st>>>(
        x, fidx, nullptr, ox, Min, S, 1, 3, 3, 0, BB);
    k_ball_query<<<blk((long long)BB * S), 256, 0, st>>>(
        x, ox, gidx, Min, S, K, radius * radius, BB);
    k_gather_group<<<blk((long long)BB * S * K * 3), 256, 0, st>>>(
        x, gidx, ox, grp, Min, S, K, 3, rowC, 0, BB);
    k_gather_group<<<blk((long long)BB * S * K * Cin), 256, 0, st>>>(
        f, gidx, nullptr, grp, Min, S, K, Cin, rowC, 3, BB);
    mlp_run(st, ar, grp, BB * S * K, Ws, dims, nw, feat);
    k_maxpool<<<blk((long long)BB * S * Cout), 256, 0, st>>>(
        feat, of, (long long)BB * S, K, Cout);
    ar.reset(m0);
}

struct PtW { const float *Wq, *Wk, *Wv, *Wp1, *bp1, *Wp2, *bp2, *Wg1, *bg1, *Wg2, *bg2, *Wo; };

void pt_block(hipStream_t st, Arena& ar, const float* x, float* f, int S, int C, const PtW& p)
{
    size_t m0 = ar.mark();
    const long long R  = (long long)BB * S;
    const long long Rp = (long long)BB * S * S;
    float* Q   = ar.get<float>(R * C);
    float* Kt  = ar.get<float>(R * C);
    float* V   = ar.get<float>(R * C);
    float* rel = ar.get<float>(Rp * 3);
    float* ph  = ar.get<float>(Rp * C);
    float* pos = ar.get<float>(Rp * C);
    float* g   = ar.get<float>(Rp * C);
    float* o   = ar.get<float>(R * C);

    gemm(st, f, p.Wq, nullptr, nullptr, Q,  (int)R, C, C, false);
    gemm(st, f, p.Wk, nullptr, nullptr, Kt, (int)R, C, C, false);
    gemm(st, f, p.Wv, nullptr, nullptr, V,  (int)R, C, C, false);
    k_rel<<<blk(Rp * 3), 256, 0, st>>>(x, rel, BB, S);
    gemm(st, rel, p.Wp1, p.bp1, nullptr, ph,  (int)Rp, C, 3, true);
    gemm(st, ph,  p.Wp2, p.bp2, nullptr, pos, (int)Rp, C, C, false);
    k_build_g<<<blk(Rp * C), 256, 0, st>>>(Q, Kt, pos, g, BB, S, C);
    gemm(st, g,  p.Wg1, p.bg1, nullptr, ph, (int)Rp, C, C, true);  // reuse ph
    gemm(st, ph, p.Wg2, p.bg2, nullptr, g,  (int)Rp, C, C, false); // overwrite g
    k_softmax_mid<<<blk(R * C), 256, 0, st>>>(g, BB, S, C);
    k_attn_out<<<blk(R * C), 256, 0, st>>>(g, pos, V, o, BB, S, C);
    gemm(st, o, p.Wo, nullptr, f, f, (int)R, C, C, false);         // + residual f
    ar.reset(m0);
}

void set_upconv(hipStream_t st, Arena& ar, const float* x1, int S1p,
                const float* x2, int S2p, const float* f1, int C1,
                const float* f2, int C2, int K,
                const float* const* Ws, const int* dims, int nw,
                const float* const* Ws2, const int* dims2, int nw2, float* out)
{
    size_t m0 = ar.mark();
    int* idx = ar.get<int>((size_t)BB * S1p * K);
    const int rowC = C2 + 3;
    float* grp = ar.get<float>((size_t)BB * S1p * K * rowC);
    k_knn<<<blk((long long)BB * S1p), 256, 0, st>>>(x1, x2, idx, S1p, S2p, K, BB);
    k_gather_group<<<blk((long long)BB * S1p * K * C2), 256, 0, st>>>(
        f2, idx, nullptr, grp, S2p, S1p, K, C2, rowC, 0, BB);
    k_gather_group<<<blk((long long)BB * S1p * K * 3), 256, 0, st>>>(
        x2, idx, x1, grp, S2p, S1p, K, 3, rowC, C2, BB);

    const float* mo = grp;
    int Cm = rowC;
    if (nw > 0) {
        Cm = dims[nw];
        float* mot = ar.get<float>((size_t)BB * S1p * K * Cm);
        mlp_run(st, ar, grp, BB * S1p * K, Ws, dims, nw, mot);
        mo = mot;
    }
    float* pooled = ar.get<float>((size_t)BB * S1p * Cm);
    k_maxpool<<<blk((long long)BB * S1p * Cm), 256, 0, st>>>(mo, pooled, (long long)BB * S1p, K, Cm);
    const int catC = Cm + C1;
    float* cat = ar.get<float>((size_t)BB * S1p * catC);
    k_scatter_ch<<<blk((long long)BB * S1p * Cm), 256, 0, st>>>(pooled, cat, (long long)BB * S1p, 1, Cm, catC, 0);
    k_scatter_ch<<<blk((long long)BB * S1p * C1), 256, 0, st>>>(f1, cat, (long long)BB * S1p, 1, C1, catC, Cm);
    mlp_run(st, ar, cat, BB * S1p, Ws2, dims2, nw2, out);
    ar.reset(m0);
}

void flow_embedding(hipStream_t st, Arena& ar, const float* x1, const float* x2,
                    const float* g1, const float* g2, const float* const* feW,
                    const int* feD, const float* Wq, const float* Wk,
                    const float* Wv, const float* Wo, float* flow)
{
    size_t m0 = ar.mark();
    const int S = 256, C = 64, D = 128, K = 32;
    int* idx = ar.get<int>((size_t)BB * S * K);
    float* grp   = ar.get<float>((size_t)BB * S * K * 131);
    float* mo    = ar.get<float>((size_t)BB * S * K * 64);
    float* flow0 = ar.get<float>((size_t)BB * S * 64);
    float* Q     = ar.get<float>((size_t)BB * S * D);
    float* K2    = ar.get<float>((size_t)BB * S * D);
    float* V     = ar.get<float>((size_t)BB * S * D);
    float* att   = ar.get<float>((size_t)BB * S * S);
    float* ca1   = ar.get<float>((size_t)BB * S * D);
    float* ca    = ar.get<float>((size_t)BB * S * 64);

    k_knn<<<blk((long long)BB * S), 256, 0, st>>>(x1, x2, idx, S, S, K, BB);
    k_gather_group<<<blk((long long)BB * S * K * C), 256, 0, st>>>(
        g2, idx, nullptr, grp, S, S, K, C, 131, 0, BB);
    k_scatter_ch<<<blk((long long)BB * S * K * C), 256, 0, st>>>(
        g1, grp, (long long)BB * S, K, C, 131, 64);
    k_gather_group<<<blk((long long)BB * S * K * 3), 256, 0, st>>>(
        x2, idx, x1, grp, S, S, K, 3, 131, 128, BB);
    mlp_run(st, ar, grp, BB * S * K, feW, feD, 3, mo);
    k_maxpool<<<blk((long long)BB * S * 64), 256, 0, st>>>(mo, flow0, (long long)BB * S, K, 64);

    gemm(st, g1, Wq, nullptr, nullptr, Q,  BB * S, D, C, false);
    gemm(st, g2, Wk, nullptr, nullptr, K2, BB * S, D, C, false);
    gemm(st, g2, Wv, nullptr, nullptr, V,  BB * S, D, C, false);
    // attn = softmax(Q @ K^T / sqrt(D))  (batched per point cloud)
    gemm(st, Q, K2, nullptr, nullptr, att, S, S, D, false,
         1.f / sqrtf((float)D), /*transB=*/true, BB,
         (long long)S * D, (long long)S * D, (long long)S * S, 0);
    k_softmax_last<<<blk((long long)BB * S), 256, 0, st>>>(att, (long long)BB * S, S);
    gemm(st, att, V, nullptr, nullptr, ca1, S, D, S, false, 1.f, false, BB,
         (long long)S * S, (long long)S * D, (long long)S * D, 0);
    gemm(st, ca1, Wo, nullptr, nullptr, ca, BB * S, 64, D, false);
    k_add<<<blk((long long)BB * S * 64), 256, 0, st>>>(flow, flow0, ca, (long long)BB * S * 64);
    ar.reset(m0);
}

const int SA1D[4]  = {5, 16, 16, 32};
const int SA2D[4]  = {35, 32, 32, 64};
const int SA3D[4]  = {67, 64, 64, 128};
const int SA4D[4]  = {131, 128, 128, 256};
const int FED[4]   = {131, 64, 64, 64};
const int SU1D2[3] = {387, 128, 128};
const int SU2D[4]  = {131, 64, 64, 128};
const int SU2D2[2] = {256, 128};
const int SU3D[4]  = {131, 64, 64, 128};
const int SU3D2[2] = {160, 128};
const int FPD[3]   = {130, 128, 128};

// d_in indices (see header comment for the assumed flattening order)
enum {
    IN_XYZS = 0, IN_POINTS = 1,
    P_SA1 = 2, P_SA2 = 5, P_SA3 = 8, P_SA4 = 11,
    P_PT2 = 14, P_PT3 = 26, P_PT4 = 38,
    P_FEMLP = 50, P_FEWQ = 53, P_FEWK = 54, P_FEWV = 55, P_FEWO = 56,
    P_SU1M2 = 57, P_SU2M = 59, P_SU2M2 = 62, P_SU3M = 63, P_SU3M2 = 66,
    P_FP = 67, P_SEMW1 = 69, P_SEMW2 = 70, P_SEMB2 = 71,
    P_OFFW1 = 72, P_OFFW2 = 73, P_OFFW3 = 74
};

PtW pt_w(void* const* din, int base) {
    auto f = [&](int i) { return (const float*)din[base + i]; };
    return PtW{f(0), f(1), f(2), f(3), f(4), f(5), f(6), f(7), f(8), f(9), f(10), f(11)};
}

} // namespace

extern "C" void kernel_launch(void* const* d_in, const int* in_sizes, int n_in,
                              void* d_out_v, int out_size, void* d_ws, size_t ws_size,
                              hipStream_t stream)
{
    (void)in_sizes; (void)n_in; (void)out_size;
    hipStream_t st = stream;
    float* d_out = (float*)d_out_v;
    const float* xyzs   = (const float*)d_in[IN_XYZS];
    const float* points = (const float*)d_in[IN_POINTS];

    Arena ar(d_ws, ws_size);

    // ---------------- persistent buffers (channel-last everywhere) --------
    float* x0a  = ar.get<float>((size_t)BB * NPTS * 3);
    float* x0b  = ar.get<float>((size_t)BB * NPTS * 3);
    float* f0a  = ar.get<float>((size_t)BB * NPTS * 2);
    float* f0b  = ar.get<float>((size_t)BB * NPTS * 2);
    float* l1xa = ar.get<float>((size_t)BB * 512 * 3);
    float* l1fa = ar.get<float>((size_t)BB * 512 * 32);
    float* l1xb = ar.get<float>((size_t)BB * 512 * 3);
    float* l1fb = ar.get<float>((size_t)BB * 512 * 32);
    float* l2xa = ar.get<float>((size_t)BB * 256 * 3);
    float* l2fa = ar.get<float>((size_t)BB * 256 * 64);
    float* l2xb = ar.get<float>((size_t)BB * 256 * 3);
    float* l2fb = ar.get<float>((size_t)BB * 256 * 64);
    float* flow = ar.get<float>((size_t)BB * 256 * 64);
    float* l3x  = ar.get<float>((size_t)BB * 128 * 3);
    float* l3f  = ar.get<float>((size_t)BB * 128 * 128);
    float* l4x  = ar.get<float>((size_t)BB * 64 * 3);
    float* l4f  = ar.get<float>((size_t)BB * 64 * 256);
    float* l3new = ar.get<float>((size_t)BB * 128 * 128);
    float* l2new = ar.get<float>((size_t)BB * 256 * 128);
    float* l1new = ar.get<float>((size_t)BB * 512 * 128);
    float* l0h   = ar.get<float>((size_t)BB * NPTS * 128);

    // ---------------- input transposes ------------------------------------
    k_cf_to_cl<<<blk((long long)BB * 3 * NPTS), 256, 0, st>>>(xyzs,              x0a, BB, 3, NPTS, 2LL * 3 * NPTS);
    k_cf_to_cl<<<blk((long long)BB * 3 * NPTS), 256, 0, st>>>(xyzs + 3 * NPTS,   x0b, BB, 3, NPTS, 2LL * 3 * NPTS);
    k_cf_to_cl<<<blk((long long)BB * 2 * NPTS), 256, 0, st>>>(points,            f0a, BB, 2, NPTS, 2LL * 2 * NPTS);
    k_cf_to_cl<<<blk((long long)BB * 2 * NPTS), 256, 0, st>>>(points + 2 * NPTS, f0b, BB, 2, NPTS, 2LL * 2 * NPTS);

    const float* sa1w[3] = {(const float*)d_in[P_SA1], (const float*)d_in[P_SA1 + 1], (const float*)d_in[P_SA1 + 2]};
    const float* sa2w[3] = {(const float*)d_in[P_SA2], (const float*)d_in[P_SA2 + 1], (const float*)d_in[P_SA2 + 2]};
    const float* sa3w[3] = {(const float*)d_in[P_SA3], (const float*)d_in[P_SA3 + 1], (const float*)d_in[P_SA3 + 2]};
    const float* sa4w[3] = {(const float*)d_in[P_SA4], (const float*)d_in[P_SA4 + 1], (const float*)d_in[P_SA4 + 2]};
    const float* few[3]  = {(const float*)d_in[P_FEMLP], (const float*)d_in[P_FEMLP + 1], (const float*)d_in[P_FEMLP + 2]};
    const float* su1w2[2] = {(const float*)d_in[P_SU1M2], (const float*)d_in[P_SU1M2 + 1]};
    const float* su2w[3]  = {(const float*)d_in[P_SU2M], (const float*)d_in[P_SU2M + 1], (const float*)d_in[P_SU2M + 2]};
    const float* su2w2[1] = {(const float*)d_in[P_SU2M2]};
    const float* su3w[3]  = {(const float*)d_in[P_SU3M], (const float*)d_in[P_SU3M + 1], (const float*)d_in[P_SU3M + 2]};
    const float* su3w2[1] = {(const float*)d_in[P_SU3M2]};
    const float* fpw[2]   = {(const float*)d_in[P_FP], (const float*)d_in[P_FP + 1]};
    const PtW pt2w = pt_w(d_in, P_PT2);
    const PtW pt3w = pt_w(d_in, P_PT3);
    const PtW pt4w = pt_w(d_in, P_PT4);

    // ---------------- encoder (both clouds) -------------------------------
    set_abstraction(st, ar, x0a, f0a, NPTS, 2, 512, 1.f, 16, sa1w, SA1D, 3, l1xa, l1fa);
    set_abstraction(st, ar, x0b, f0b, NPTS, 2, 512, 1.f, 16, sa1w, SA1D, 3, l1xb, l1fb);
    set_abstraction(st, ar, l1xa, l1fa, 512, 32, 256, 2.f, 8, sa2w, SA2D, 3, l2xa, l2fa);
    set_abstraction(st, ar, l1xb, l1fb, 512, 32, 256, 2.f, 8, sa2w, SA2D, 3, l2xb, l2fb);
    pt_block(st, ar, l2xa, l2fa, 256, 64, pt2w);
    pt_block(st, ar, l2xb, l2fb, 256, 64, pt2w);

    // ---------------- flow embedding --------------------------------------
    flow_embedding(st, ar, l2xa, l2xb, l2fa, l2fb, few, FED,
                   (const float*)d_in[P_FEWQ], (const float*)d_in[P_FEWK],
                   (const float*)d_in[P_FEWV], (const float*)d_in[P_FEWO], flow);

    // ---------------- deeper encoder on cloud 1 ----------------------------
    set_abstraction(st, ar, l2xa, flow, 256, 64, 128, 4.f, 4, sa3w, SA3D, 3, l3x, l3f);
    pt_block(st, ar, l3x, l3f, 128, 128, pt3w);
    set_abstraction(st, ar, l3x, l3f, 128, 128, 64, 8.f, 2, sa4w, SA4D, 3, l4x, l4f);
    pt_block(st, ar, l4x, l4f, 64, 256, pt4w);

    // ---------------- decoder ---------------------------------------------
    set_upconv(st, ar, l3x, 128, l4x, 64, l3f, 128, l4f, 256, 8,
               nullptr, nullptr, 0, su1w2, SU1D2, 2, l3new);
    {
        size_t m0 = ar.mark();
        float* f1cat = ar.get<float>((size_t)BB * 256 * 128);
        k_scatter_ch<<<blk((long long)BB * 256 * 64), 256, 0, st>>>(l2fa, f1cat, (long long)BB * 256, 1, 64, 128, 0);
        k_scatter_ch<<<blk((long long)BB * 256 * 64), 256, 0, st>>>(flow, f1cat, (long long)BB * 256, 1, 64, 128, 64);
        set_upconv(st, ar, l2xa, 256, l3x, 128, f1cat, 128, l3new, 128, 8,
                   su2w, SU2D, 3, su2w2, SU2D2, 1, l2new);
        ar.reset(m0);
    }
    set_upconv(st, ar, l1xa, 512, l2xa, 256, l1fa, 32, l2new, 128, 8,
               su3w, SU3D, 3, su3w2, SU3D2, 1, l1new);

    // ---------------- feature propagation to full resolution ---------------
    {
        size_t m0 = ar.mark();
        float* fpin = ar.get<float>((size_t)BB * NPTS * 130);
        k_scatter_ch<<<blk((long long)BB * NPTS * 2), 256, 0, st>>>(f0a, fpin, (long long)BB * NPTS, 1, 2, 130, 0);
        k_fp_interp<<<blk((long long)BB * NPTS), 256, 0, st>>>(x0a, l1xa, l1new, fpin, BB, NPTS, 512, 128, 130, 2);
        mlp_run(st, ar, fpin, BB * NPTS, fpw, FPD, 2, l0h);
        ar.reset(m0);
    }

    // ---------------- heads -------------------------------------------------
    {
        size_t m0 = ar.mark();
        const long long R = (long long)BB * NPTS;
        float* t64 = ar.get<float>((size_t)R * 64);
        float* s2  = ar.get<float>((size_t)R * 2);
        float* o32 = ar.get<float>((size_t)R * 32);
        // sem = log_softmax(relu(h @ W1) @ W2 + b2)
        gemm(st, l0h, (const float*)d_in[P_SEMW1], nullptr, nullptr, t64, (int)R, 64, 128, true);
        gemm(st, t64, (const float*)d_in[P_SEMW2], (const float*)d_in[P_SEMB2], nullptr, s2, (int)R, 2, 64, false);
        k_logsoftmax2<<<blk(R), 256, 0, st>>>(s2, d_out, R);
        // off = relu(relu(h @ W1) @ W2) @ W3
        gemm(st, l0h, (const float*)d_in[P_OFFW1], nullptr, nullptr, t64, (int)R, 64, 128, true);
        gemm(st, t64, (const float*)d_in[P_OFFW2], nullptr, nullptr, o32, (int)R, 32, 64, true);
        gemm(st, o32, (const float*)d_in[P_OFFW3], nullptr, nullptr, d_out + R * 2, (int)R, 3, 32, false);
        ar.reset(m0);
    }
}
